// LSTMNet_10462540333634
// MI455X (gfx1250) — matrix operand order repaired and checked by reading, not compiled, not run on hardware
//
#include <hip/hip_runtime.h>
#include <hip/hip_bf16.h>

typedef __attribute__((ext_vector_type(16))) __bf16 v16bf;
typedef __attribute__((ext_vector_type(8)))  __bf16 v8bf;
typedef __attribute__((ext_vector_type(8)))  float  v8f;

#define T_STEPS 512
#define BSZ     64
#define IDIM    512
#define HDIM    1024
#define ODIM    512
#define GDIM    4096   // 4*HDIM
#define NBLK    64     // persistent grid

// ---- global workspace layout (bytes) ----
#define OFF_XBF   0u                         // [T,B,I] bf16 : 33,554,432
#define OFF_WIH   33554432u                  // [4H,I]  bf16 :  4,194,304
#define OFF_WHH   37748736u                  // [4H,H]  bf16 :  8,388,608
#define OFF_WHO   46137344u                  // [O,H]   bf16 :  1,048,576
#define OFF_BSUM  47185920u                  // [4H]    f32  :     16,384
#define OFF_HST   47202304u                  // [B,H]   bf16 :    131,072
#define OFF_BAR   47333376u                  // barrier counter
#define OFF_XP    50331648u                  // [T*B,4H] f32 (path A only)
#define XP_BYTES  536870912ull               // 512 MB

// ---- dynamic LDS layouts (bytes) ----
// path A (precomputed Xp):
#define LA_GOFF   0u        // gates [4][64][16] f32 : 16384
#define LA_COFF   16384u    // c     [64][16]    f32 :  4096
#define LA_XPOFF  20480u    // xp    [2][4][64][16] f32 : 2*16384
#define LA_TOTAL  53248u
// path B (fused input projection):
#define LB_GOFF   0u        // gates [4][64][16] f32 : 16384
#define LB_COFF   16384u    // c     [64][16]    f32 :  4096
#define LB_XOFF   20480u    // xbuf  [2][64][512] bf16 : 2*65536
#define LB_TOTAL  151552u

__device__ __forceinline__ float sigm(float x) {
    return 1.0f / (1.0f + __expf(-x));
}
__device__ __forceinline__ float tanh_fast(float x) {
    float e = __expf(2.0f * x);
    return (e - 1.0f) / (e + 1.0f);
}

// A-fragment for v_wmma_f32_16x16x32_bf16 (lane=M, VGPRs pack K)
__device__ __forceinline__ v16bf load_a_frag(const __bf16* row, int k0, int half) {
    v8bf lo = *reinterpret_cast<const v8bf*>(row + k0 + half * 8);
    v8bf hi = *reinterpret_cast<const v8bf*>(row + k0 + 16 + half * 8);
    return __builtin_shufflevector(lo, hi, 0,1,2,3,4,5,6,7,8,9,10,11,12,13,14,15);
}
// B-fragment (lane=N, VGPRs pack K): W[n, k0+half*8 ..+8] then W[n, k0+16+half*8 ..+8]
__device__ __forceinline__ v16bf load_b_frag(const __bf16* __restrict__ wrow,
                                             int k0, int half) {
    v8bf lo = *reinterpret_cast<const v8bf*>(wrow + k0 + half * 8);
    v8bf hi = *reinterpret_cast<const v8bf*>(wrow + k0 + 16 + half * 8);
    return __builtin_shufflevector(lo, hi, 0,1,2,3,4,5,6,7,8,9,10,11,12,13,14,15);
}

// CDNA5 async global->LDS copy (ASYNCcnt-tracked), 16 bytes per lane.
__device__ __forceinline__ void async_ld_b128(unsigned lds_byte_off, const void* gaddr) {
    asm volatile("global_load_async_to_lds_b128 %0, %1, off"
                 :: "v"(lds_byte_off), "v"(gaddr) : "memory");
}
__device__ __forceinline__ void wait_async0() {
    asm volatile("s_wait_asynccnt 0x0" ::: "memory");
}

__device__ __forceinline__ void grid_barrier(unsigned int* bar, unsigned int target) {
    __threadfence();
    __syncthreads();
    if (threadIdx.x == 0) {
        __hip_atomic_fetch_add(bar, 1u, __ATOMIC_ACQ_REL, __HIP_MEMORY_SCOPE_AGENT);
        while (__hip_atomic_load(bar, __ATOMIC_ACQUIRE, __HIP_MEMORY_SCOPE_AGENT) < target) {
            __builtin_amdgcn_s_sleep(2);
        }
    }
    __syncthreads();
}

// ---------------- prep kernels ----------------
__global__ void cvt_f32_bf16(const float* __restrict__ src, __bf16* __restrict__ dst, int n) {
    int i = blockIdx.x * blockDim.x + threadIdx.x;
    if (i < n) dst[i] = (__bf16)src[i];
}
__global__ void bias_sum(const float* __restrict__ a, const float* __restrict__ b,
                         float* __restrict__ o, int n) {
    int i = blockIdx.x * blockDim.x + threadIdx.x;
    if (i < n) o[i] = a[i] + b[i];
}

// ---------------- path A: big parallel GEMM  Xp = X @ Wih^T + bsum ----------------
// M=32768 (T*B), N=4096, K=512. Wave: 1 N-tile x 4 M-tiles of a 64-row group.
__global__ void __launch_bounds__(256)
gemm_xp(const __bf16* __restrict__ Xbf,   // [T*B, I]
        const __bf16* __restrict__ Wih,   // [4H, I]
        const float*  __restrict__ bsum,  // [4H]
        float* __restrict__ Xp) {         // [T*B, 4H]
    const int tid  = threadIdx.x;
    const int lane = tid & 31;
    const int half = lane >> 4;
    const int row  = lane & 15;
    const int gw   = blockIdx.x * 8 + (tid >> 5);   // 0 .. 131071
    const int nt   = gw & 255;
    const int mg   = gw >> 8;                       // 64-row group (== timestep)
    const int n    = nt * 16 + row;
    const int rbase = mg * 64;
    const __bf16* wrow = Wih + (size_t)n * IDIM;

    v8f acc[4];
    #pragma unroll
    for (int mt = 0; mt < 4; ++mt) acc[mt] = (v8f){0.f,0.f,0.f,0.f,0.f,0.f,0.f,0.f};

    for (int k0 = 0; k0 < IDIM; k0 += 32) {
        v16bf b = load_b_frag(wrow, k0, half);
        #pragma unroll
        for (int mt = 0; mt < 4; ++mt) {
            v16bf a = load_a_frag(Xbf + (size_t)(rbase + mt * 16 + row) * IDIM, k0, half);
            acc[mt] = __builtin_amdgcn_wmma_f32_16x16x32_bf16(
                false, a, false, b, (short)0, acc[mt], false, false);
        }
    }
    const float bias = bsum[n];
    #pragma unroll
    for (int mt = 0; mt < 4; ++mt) {
        #pragma unroll
        for (int r = 0; r < 8; ++r) {
            Xp[(size_t)(rbase + mt * 16 + r + half * 8) * GDIM + n] = acc[mt][r] + bias;
        }
    }
}

// ---------------- path A: persistent recurrence (K = HDIM only) ----------------
// 64 blocks; block owns hd cols [blockIdx*16,+16). wave w: gate g=w&3, M-half
// msel=w>>2 (2 M-tiles). Gates + c in LDS; Xp slice async double-buffered.
__global__ void __launch_bounds__(256)
lstm_persistent_xp(const float*  __restrict__ Xp,   // [T*B,4H], bias included
                   const __bf16* __restrict__ Whh,  // [4H,H]
                   __bf16* __restrict__ Hst,        // [B,H]
                   unsigned int* bar) {
    extern __shared__ char smem[];
    float* lgate = (float*)(smem + LA_GOFF);          // [4][64][16]
    float* lc    = (float*)(smem + LA_COFF);          // [64][16]
    const float* lxp = (const float*)(smem + LA_XPOFF); // [2][4][64][16]

    const int tid  = threadIdx.x;
    const int lane = tid & 31;
    const int half = lane >> 4;
    const int row  = lane & 15;
    const int wave = tid >> 5;
    const int g    = wave & 3;
    const int msel = wave >> 2;                        // 0..1
    const int n    = (g * 64 + blockIdx.x) * 16 + row; // gate column
    const __bf16* wrow_h = Whh + (size_t)n * HDIM;
    const int hbase = blockIdx.x * 16;
    const unsigned dynbase = __builtin_amdgcn_groupstaticsize();

    for (int e = tid; e < BSZ * 16; e += 256) lc[e] = 0.0f;

    // stage Xp slice for t=0: 16KB = 1024 x 16B chunks, 4 per thread
    #pragma unroll
    for (int j = 0; j < 4; ++j) {
        int c = j * 256 + tid;
        int gc = c >> 8, bc = (c >> 2) & 63, q = c & 3;
        const float* ga = Xp + (size_t)bc * GDIM + gc * 1024 + hbase + q * 4;
        async_ld_b128(dynbase + LA_XPOFF + ((gc * 64 + bc) * 16 + q * 4) * 4u, ga);
    }

    for (int t = 0; t < T_STEPS; ++t) {
        wait_async0();
        __syncthreads();

        if (t + 1 < T_STEPS) {
            unsigned lb = dynbase + LA_XPOFF + ((unsigned)(t + 1) & 1u) * 16384u;
            const float* Xn = Xp + (size_t)(t + 1) * (BSZ * GDIM);
            #pragma unroll
            for (int j = 0; j < 4; ++j) {
                int c = j * 256 + tid;
                int gc = c >> 8, bc = (c >> 2) & 63, q = c & 3;
                async_ld_b128(lb + ((gc * 64 + bc) * 16 + q * 4) * 4u,
                              Xn + (size_t)bc * GDIM + gc * 1024 + hbase + q * 4);
            }
        }

        // accumulator init = Xp slice (bias pre-added)
        const float* xps = lxp + ((unsigned)t & 1u) * 4096u;
        v8f acc[2];
        #pragma unroll
        for (int mi = 0; mi < 2; ++mi) {
            int mt = msel * 2 + mi;
            #pragma unroll
            for (int r = 0; r < 8; ++r)
                acc[mi][r] = xps[(g * 64 + mt * 16 + r + half * 8) * 16 + row];
        }
        // recurrent GEMM: K = HDIM
        for (int k0 = 0; k0 < HDIM; k0 += 32) {
            __builtin_prefetch(wrow_h + k0 + 256, 0, 3);
            v16bf b = load_b_frag(wrow_h, k0, half);
            #pragma unroll
            for (int mi = 0; mi < 2; ++mi) {
                int mt = msel * 2 + mi;
                v16bf a = load_a_frag(Hst + (size_t)(mt * 16 + row) * HDIM, k0, half);
                acc[mi] = __builtin_amdgcn_wmma_f32_16x16x32_bf16(
                    false, a, false, b, (short)0, acc[mi], false, false);
            }
        }
        #pragma unroll
        for (int mi = 0; mi < 2; ++mi) {
            int mt = msel * 2 + mi;
            #pragma unroll
            for (int r = 0; r < 8; ++r)
                lgate[(g * 64 + mt * 16 + r + half * 8) * 16 + row] = acc[mi][r];
        }
        __syncthreads();

        for (int e = tid; e < BSZ * 16; e += 256) {
            int b  = e >> 4;
            int hl = e & 15;
            float gi = lgate[(0 * 64 + b) * 16 + hl];
            float gf = lgate[(1 * 64 + b) * 16 + hl];
            float gg = lgate[(2 * 64 + b) * 16 + hl];
            float go = lgate[(3 * 64 + b) * 16 + hl];
            float c  = lc[e];
            c = sigm(gf) * c + sigm(gi) * tanh_fast(gg);
            float h = sigm(go) * tanh_fast(c);
            lc[e] = c;
            Hst[(size_t)b * HDIM + hbase + hl] = (__bf16)h;
        }

        grid_barrier(bar, (unsigned int)(t + 1) * NBLK);
    }
}

// ---------------- path B: fused persistent recurrence (fallback) ----------------
__global__ void __launch_bounds__(256)
lstm_persistent_fused(const __bf16* __restrict__ Xbf,  // [T,B,I]
                      const __bf16* __restrict__ Wih,  // [4H,I]
                      const __bf16* __restrict__ Whh,  // [4H,H]
                      const float*  __restrict__ bsum, // [4H]
                      __bf16* __restrict__ Hst,        // [B,H]
                      unsigned int* bar) {
    extern __shared__ char smem[];
    float* lgate = (float*)(smem + LB_GOFF);   // [4][64][16]
    float* lc    = (float*)(smem + LB_COFF);   // [64][16]

    const int tid  = threadIdx.x;
    const int lane = tid & 31;
    const int half = lane >> 4;
    const int row  = lane & 15;
    const int wave = tid >> 5;
    const int g    = wave & 3;
    const int msel = wave >> 2;
    const int n    = (g * 64 + blockIdx.x) * 16 + row;
    const float bias = bsum[n];
    const __bf16* wrow_i = Wih + (size_t)n * IDIM;
    const __bf16* wrow_h = Whh + (size_t)n * HDIM;
    const int hbase = blockIdx.x * 16;
    const unsigned dynbase = __builtin_amdgcn_groupstaticsize();

    for (int e = tid; e < BSZ * 16; e += 256) lc[e] = 0.0f;

    // stage x_0: 64KB = 4096 x 16B chunks, 16 per thread
    #pragma unroll
    for (int j = 0; j < 16; ++j) {
        int chunk = j * 256 + tid;
        async_ld_b128(dynbase + LB_XOFF + chunk * 16u, Xbf + (size_t)chunk * 8);
    }

    for (int t = 0; t < T_STEPS; ++t) {
        wait_async0();
        __syncthreads();

        if (t + 1 < T_STEPS) {
            const __bf16* Xn = Xbf + (size_t)(t + 1) * (BSZ * IDIM);
            unsigned lb = dynbase + LB_XOFF + ((unsigned)(t + 1) & 1u) * 65536u;
            #pragma unroll
            for (int j = 0; j < 16; ++j) {
                int chunk = j * 256 + tid;
                async_ld_b128(lb + chunk * 16u, Xn + (size_t)chunk * 8);
            }
        }

        const __bf16* xb = (const __bf16*)(smem + LB_XOFF + ((unsigned)t & 1u) * 65536u);
        v8f acc[2];
        #pragma unroll
        for (int mi = 0; mi < 2; ++mi) acc[mi] = (v8f){0.f,0.f,0.f,0.f,0.f,0.f,0.f,0.f};

        for (int k0 = 0; k0 < IDIM; k0 += 32) {
            __builtin_prefetch(wrow_i + k0 + 256, 0, 3);
            v16bf b = load_b_frag(wrow_i, k0, half);
            #pragma unroll
            for (int mi = 0; mi < 2; ++mi) {
                int mt = msel * 2 + mi;
                v16bf a = load_a_frag(xb + (size_t)(mt * 16 + row) * IDIM, k0, half);
                acc[mi] = __builtin_amdgcn_wmma_f32_16x16x32_bf16(
                    false, a, false, b, (short)0, acc[mi], false, false);
            }
        }
        for (int k0 = 0; k0 < HDIM; k0 += 32) {
            __builtin_prefetch(wrow_h + k0 + 256, 0, 3);
            v16bf b = load_b_frag(wrow_h, k0, half);
            #pragma unroll
            for (int mi = 0; mi < 2; ++mi) {
                int mt = msel * 2 + mi;
                v16bf a = load_a_frag(Hst + (size_t)(mt * 16 + row) * HDIM, k0, half);
                acc[mi] = __builtin_amdgcn_wmma_f32_16x16x32_bf16(
                    false, a, false, b, (short)0, acc[mi], false, false);
            }
        }
        #pragma unroll
        for (int mi = 0; mi < 2; ++mi) {
            int mt = msel * 2 + mi;
            #pragma unroll
            for (int r = 0; r < 8; ++r)
                lgate[(g * 64 + mt * 16 + r + half * 8) * 16 + row] = acc[mi][r] + bias;
        }
        __syncthreads();

        for (int e = tid; e < BSZ * 16; e += 256) {
            int b  = e >> 4;
            int hl = e & 15;
            float gi = lgate[(0 * 64 + b) * 16 + hl];
            float gf = lgate[(1 * 64 + b) * 16 + hl];
            float gg = lgate[(2 * 64 + b) * 16 + hl];
            float go = lgate[(3 * 64 + b) * 16 + hl];
            float c  = lc[e];
            c = sigm(gf) * c + sigm(gi) * tanh_fast(gg);
            float h = sigm(go) * tanh_fast(c);
            lc[e] = c;
            Hst[(size_t)b * HDIM + hbase + hl] = (__bf16)h;
        }

        grid_barrier(bar, (unsigned int)(t + 1) * NBLK);
    }
}

// ---------------- output head: out = h @ W_ho^T + b_ho ----------------
__global__ void __launch_bounds__(256)
out_gemm(const __bf16* __restrict__ Hst, const __bf16* __restrict__ Who,
         const float* __restrict__ b_ho, float* __restrict__ out) {
    const int tid  = threadIdx.x;
    const int lane = tid & 31;
    const int half = lane >> 4;
    const int row  = lane & 15;
    const int nt   = blockIdx.x * 8 + (tid >> 5);   // 0..31
    const int n    = nt * 16 + row;
    const __bf16* wrow = Who + (size_t)n * HDIM;

    v8f acc[4];
    #pragma unroll
    for (int mt = 0; mt < 4; ++mt) acc[mt] = (v8f){0.f,0.f,0.f,0.f,0.f,0.f,0.f,0.f};

    for (int k0 = 0; k0 < HDIM; k0 += 32) {
        v16bf b = load_b_frag(wrow, k0, half);
        #pragma unroll
        for (int mt = 0; mt < 4; ++mt) {
            v16bf a = load_a_frag(Hst + (size_t)(mt * 16 + row) * HDIM, k0, half);
            acc[mt] = __builtin_amdgcn_wmma_f32_16x16x32_bf16(
                false, a, false, b, (short)0, acc[mt], false, false);
        }
    }
    const float bias = b_ho[n];
    #pragma unroll
    for (int mt = 0; mt < 4; ++mt) {
        #pragma unroll
        for (int r = 0; r < 8; ++r)
            out[(size_t)(mt * 16 + r + half * 8) * ODIM + n] = acc[mt][r] + bias;
    }
}

extern "C" void kernel_launch(void* const* d_in, const int* in_sizes, int n_in,
                              void* d_out, int out_size, void* d_ws, size_t ws_size,
                              hipStream_t stream) {
    const float* X    = (const float*)d_in[0];
    const float* W_ih = (const float*)d_in[1];
    const float* W_hh = (const float*)d_in[2];
    const float* b_ih = (const float*)d_in[3];
    const float* b_hh = (const float*)d_in[4];
    const float* W_ho = (const float*)d_in[5];
    const float* b_ho = (const float*)d_in[6];
    float* out = (float*)d_out;

    char* ws = (char*)d_ws;
    __bf16*       Xbf  = (__bf16*)(ws + OFF_XBF);
    __bf16*       Wihb = (__bf16*)(ws + OFF_WIH);
    __bf16*       Whhb = (__bf16*)(ws + OFF_WHH);
    __bf16*       Whob = (__bf16*)(ws + OFF_WHO);
    float*        bsm  = (float*)(ws + OFF_BSUM);
    __bf16*       Hst  = (__bf16*)(ws + OFF_HST);
    unsigned int* bar  = (unsigned int*)(ws + OFF_BAR);
    float*        Xp   = (float*)(ws + OFF_XP);

    hipMemsetAsync(Hst, 0, (size_t)BSZ * HDIM * sizeof(__bf16), stream);
    hipMemsetAsync(bar, 0, 128, stream);

    const int nx   = T_STEPS * BSZ * IDIM;
    const int nwih = GDIM * IDIM;
    const int nwhh = GDIM * HDIM;
    const int nwho = ODIM * HDIM;
    cvt_f32_bf16<<<(nx   + 255) / 256, 256, 0, stream>>>(X,    Xbf,  nx);
    cvt_f32_bf16<<<(nwih + 255) / 256, 256, 0, stream>>>(W_ih, Wihb, nwih);
    cvt_f32_bf16<<<(nwhh + 255) / 256, 256, 0, stream>>>(W_hh, Whhb, nwhh);
    cvt_f32_bf16<<<(nwho + 255) / 256, 256, 0, stream>>>(W_ho, Whob, nwho);
    bias_sum<<<(GDIM + 255) / 256, 256, 0, stream>>>(b_ih, b_hh, bsm, GDIM);

    const bool pathA = ws_size >= (size_t)OFF_XP + XP_BYTES;
    if (pathA) {
        // full-GPU input-projection GEMM, then shortened serial recurrence
        gemm_xp<<<16384, 256, 0, stream>>>(Xbf, Wihb, bsm, Xp);
        lstm_persistent_xp<<<NBLK, 256, LA_TOTAL, stream>>>(Xp, Whhb, Hst, bar);
    } else {
        lstm_persistent_fused<<<NBLK, 256, LB_TOTAL, stream>>>(Xbf, Wihb, Whhb, bsm, Hst, bar);
    }
    out_gemm<<<4, 256, 0, stream>>>(Hst, Whob, b_ho, out);
}
